// TensorTrainGaussian_42305427866351
// MI455X (gfx1250) — matrix operand order repaired and checked by reading, not compiled, not run on hardware
//
#include <hip/hip_runtime.h>
#include <hip/hip_bf16.h>
#include <stdint.h>

typedef _Float16 f16;
typedef _Float16 v16h __attribute__((ext_vector_type(16)));
typedef _Float16 v8h  __attribute__((ext_vector_type(8)));
typedef float    v8f  __attribute__((ext_vector_type(8)));

#define NN 32768
#define MM 16
#define KK 64
#define HALF_LOG_2PI 0.9189385332046727f
#define EPS_F 2.220446049250313e-16f

// workspace layout (in floats)
#define WS_WK0 0
#define WS_LSE 64
#define WS_PK  (64 + 1024)        // 16 steps * 12288 floats: [coef|invs|mu] per step
#define PK_STRIDE 12288

// ---------------- CDNA5 async global->LDS helpers ----------------

// One wave-level async DMA: each lane moves 16B global -> LDS (ASYNCcnt).
// GVS mode: addr = SADDR(64-bit sgpr pair) + VADDR(32-bit byte offset).
__device__ __forceinline__ void async_b128(uint32_t lds_byte, uint32_t gbyte,
                                           const float* gbase) {
    asm volatile("global_load_async_to_lds_b128 %0, %1, %2"
                 :: "v"(lds_byte), "v"(gbyte), "s"(gbase)
                 : "memory");
}

// Stage one 48KB packed parameter block (12288 floats) into LDS.
// 12 issues/wave * 32 lanes * 16B = 6KB/wave; 8 waves cover 48KB.
__device__ __forceinline__ void stage48k_async(uint32_t lds_base_byte,
                                               const float* g) {
    const uint32_t t16 = threadIdx.x * 16u;
    #pragma unroll
    for (int c = 0; c < 12; ++c) {
        async_b128(lds_base_byte + t16 + (uint32_t)c * 4096u,
                   t16 + (uint32_t)c * 4096u, g);
    }
}

__device__ __forceinline__ void wait_async0() {
#if defined(__has_builtin)
#if __has_builtin(__builtin_amdgcn_s_wait_asynccnt)
    __builtin_amdgcn_s_wait_asynccnt(0);
#else
    asm volatile("s_wait_asynccnt 0" ::: "memory");
#endif
#else
    asm volatile("s_wait_asynccnt 0" ::: "memory");
#endif
}

// ---------------- precompute kernels ----------------

// logsumexp over axis=1 (rows i) of W_logits[m,:,j] -> lse[m*64+j]
__global__ void prep_lse(const float* __restrict__ Wl, float* __restrict__ lse) {
    int gid = blockIdx.x * blockDim.x + threadIdx.x;
    if (gid >= MM * KK) return;
    int m = gid >> 6, j = gid & 63;
    const float* base = Wl + m * KK * KK + j;
    float mx = -3.4e38f;
    for (int i = 0; i < KK; ++i) mx = fmaxf(mx, base[i * KK]);
    float s = 0.f;
    for (int i = 0; i < KK; ++i) s += __expf(base[i * KK] - mx);
    lse[gid] = mx + __logf(s);
}

// wk0 = softmax(wk0_logits[0,:])
__global__ void prep_wk0(const float* __restrict__ logits, float* __restrict__ wk0) {
    __shared__ float sm[64];
    int t = threadIdx.x;
    float v = logits[t];
    sm[t] = v; __syncthreads();
    for (int off = 32; off > 0; off >>= 1) {
        if (t < off) sm[t] = fmaxf(sm[t], sm[t + off]);
        __syncthreads();
    }
    float mx = sm[0]; __syncthreads();
    float e = __expf(v - mx);
    sm[t] = e; __syncthreads();
    for (int off = 32; off > 0; off >>= 1) {
        if (t < off) sm[t] += sm[t + off];
        __syncthreads();
    }
    wk0[t] = e / sm[0];
}

// Pack per-step params contiguously: pk[m] = [coef(4096) | invs(4096) | mu(4096)]
// coef = logW - log(sigma) - 0.5*log(2*pi), invs = 1/sigma
__global__ void prep_pack(const float* __restrict__ Wl, const float* __restrict__ psig,
                          const float* __restrict__ mug, const float* __restrict__ lse,
                          float* __restrict__ pk) {
    int idx = blockIdx.x * blockDim.x + threadIdx.x;
    if (idx >= MM * KK * KK) return;
    int m = idx >> 12, e = idx & 4095, j = idx & 63;
    float p  = psig[idx];
    float sg = (p > 20.f) ? p : log1pf(__expf(p));   // softplus
    float* dst = pk + m * PK_STRIDE;
    dst[e]        = Wl[idx] - lse[(m << 6) | j] - __logf(sg) - HALF_LOG_2PI;
    dst[4096 + e] = 1.0f / sg;
    dst[8192 + e] = mug[idx];
}

// ---------------- main chained-matmul kernel ----------------
// 256 threads = 8 wave32s: waves 0-3 -> sample 2*blockIdx.x, waves 4-7 -> +1.
// Each wave owns a 16x64 row stripe of prod; per step: 2 K-chunks x 4 col tiles
// of v_wmma_f32_16x16x32_f16. Params double-buffered via async global->LDS DMA.

__global__ __launch_bounds__(256) void tt_main(
    const float* __restrict__ X, const float* __restrict__ pk,
    const float* __restrict__ wk0, float* __restrict__ out) {

    __shared__ alignas(16) float spbuf[2][PK_STRIDE];  // double-buffered params, 96KB
    __shared__ alignas(16) f16 resultH[2][KK * KK];    // per-sample B (row-major result)
    __shared__ alignas(16) f16 prodH[2][KK * KK];      // per-sample A (f16 copy of prod)
    __shared__ float sp_wk0[KK];
    __shared__ float red[256];

    const int tid  = threadIdx.x;
    const int lane = tid & 31;
    const int wave = tid >> 5;
    const int s    = wave >> 2;      // sample slot 0/1
    const int wsub = wave & 3;       // stripe index within sample
    const int lt   = tid & 127;      // thread id within sample group
    const int n    = blockIdx.x * 2 + s;

    // LDS byte addresses of the two param buffers (generic ptr low 32 bits)
    const uint32_t lds_buf0 = (uint32_t)(uintptr_t)&spbuf[0][0];
    const uint32_t lds_buf1 = (uint32_t)(uintptr_t)&spbuf[1][0];

    // prefetch step 0 params
    stage48k_async(lds_buf0, pk);

    // prod = Identity (f16)
    for (int idx = tid; idx < 2 * KK * KK; idx += 256) {
        int slot = idx >> 12, e = idx & 4095;
        prodH[slot][e] = (f16)(((e >> 6) == (e & 63)) ? 1.0f : 0.0f);
    }
    if (tid < KK) sp_wk0[tid] = wk0[tid];

    const int r0  = wsub * 16;
    const int row = lane & 15;
    const int hi  = lane >> 4;

    const v8f vzero = {0.f, 0.f, 0.f, 0.f, 0.f, 0.f, 0.f, 0.f};
    v8f acc[4];

    for (int m = 0; m < MM; ++m) {
        const int cur = m & 1;
        wait_async0();     // this wave's staged params have landed
        __syncthreads();   // everyone's params landed; prior-step consumers done

        // prefetch next step into the other buffer (its readers finished
        // two barriers ago)
        if (m + 1 < MM) {
            stage48k_async(cur ? lds_buf0 : lds_buf1, pk + (m + 1) * PK_STRIDE);
        }

        // elementwise: result[k][j] = exp(coef - 0.5*((x-mu)*invs)^2), f16,
        // 8 elements per thread per iteration -> ds_store_b128
        const float* sp = spbuf[cur];
        const float x = X[n * MM + m];
        #pragma unroll
        for (int v = 0; v < 4; ++v) {
            const int e0 = (v * 128 + lt) * 8;
            union { v8h h; f16 e[8]; } pack;
            #pragma unroll
            for (int q = 0; q < 8; ++q) {
                int e = e0 + q;
                float z = (x - sp[8192 + e]) * sp[4096 + e];
                pack.e[q] = (f16)__expf(sp[e] - 0.5f * z * z);
            }
            *(v8h*)&resultH[s][e0] = pack.h;
        }
        __syncthreads();

        // matmul: new_prod stripe = prod_stripe @ result^T
        acc[0] = vzero; acc[1] = vzero; acc[2] = vzero; acc[3] = vzero;
        #pragma unroll
        for (int jt = 0; jt < 2; ++jt) {
            // A fragment: rows r0..r0+15, K-chunk jt*32..+31.
            // lane<16 holds K 0-7 & 16-23, lane>=16 holds K 8-15 & 24-31.
            const f16* pa = &prodH[s][(r0 + row) * KK + jt * 32 + hi * 8];
            union { v8h h[2]; v16h v; } ua;
            ua.h[0] = *(const v8h*)pa;
            ua.h[1] = *(const v8h*)(pa + 16);
            const v16h a = ua.v;
            #pragma unroll
            for (int kt = 0; kt < 4; ++kt) {
                // B fragment: B[j,k] = result[k,j]; lane = column k,
                // j halves jt*32 + hi*8 + 0-7 and + 16-23 (same K order as A).
                const f16* pb = &resultH[s][(kt * 16 + row) * KK + jt * 32 + hi * 8];
                union { v8h h[2]; v16h v; } ub;
                ub.h[0] = *(const v8h*)pb;
                ub.h[1] = *(const v8h*)(pb + 16);
                const v16h b = ub.v;
                acc[kt] = __builtin_amdgcn_wmma_f32_16x16x32_f16(
                    false, a, false, b, (short)0, acc[kt], false, false);
            }
        }
        __syncthreads();  // all reads of prodH/resultH done before overwrite

        if (m < MM - 1) {
            // C/D layout: lane = column (N), VGPR r = row M (lane>=16 -> +8)
            #pragma unroll
            for (int kt = 0; kt < 4; ++kt) {
                #pragma unroll
                for (int r = 0; r < 8; ++r) {
                    prodH[s][(r0 + hi * 8 + r) * KK + kt * 16 + (lane & 15)] =
                        (f16)acc[kt][r];
                }
            }
        }
    }

    // likelihood = sum_{k,j} wk0[k] * prod[k,j], from last-step fp32 accumulators
    float part = 0.f;
    #pragma unroll
    for (int kt = 0; kt < 4; ++kt) {
        #pragma unroll
        for (int r = 0; r < 8; ++r) {
            part += sp_wk0[r0 + hi * 8 + r] * acc[kt][r];
        }
    }
    red[tid] = part;
    __syncthreads();
    for (int off = 64; off > 0; off >>= 1) {
        if (lt < off) red[tid] += red[tid + off];
        __syncthreads();
    }
    if (lt == 0) out[n] = __logf(red[tid] + EPS_F);
}

// ---------------- launch ----------------

extern "C" void kernel_launch(void* const* d_in, const int* in_sizes, int n_in,
                              void* d_out, int out_size, void* d_ws, size_t ws_size,
                              hipStream_t stream) {
    const float* X   = (const float*)d_in[0];  // (N, M)
    const float* wl0 = (const float*)d_in[1];  // (1, K)
    const float* Wl  = (const float*)d_in[2];  // (M, K, K)
    const float* mu  = (const float*)d_in[3];  // (M, K, K)
    const float* ps  = (const float*)d_in[4];  // (M, K, K)
    float* out = (float*)d_out;

    float* ws  = (float*)d_ws;
    float* wk0 = ws + WS_WK0;
    float* lse = ws + WS_LSE;
    float* pk  = ws + WS_PK;

    prep_lse <<<(MM * KK + 255) / 256, 256, 0, stream>>>(Wl, lse);
    prep_wk0 <<<1, 64, 0, stream>>>(wl0, wk0);
    prep_pack<<<(MM * KK * KK) / 256, 256, 0, stream>>>(Wl, ps, mu, lse, pk);
    tt_main  <<<NN / 2, 256, 0, stream>>>(X, pk, wk0, out);
}
